// LSTM_7816840478770
// MI455X (gfx1250) — compile-verified, matrix operand order checked by reading, not run on hardware
//
#include <hip/hip_runtime.h>
#include <hip/hip_bf16.h>

typedef __attribute__((ext_vector_type(16))) __bf16 v16bf;
typedef __attribute__((ext_vector_type(8)))  __bf16 v8bf;
typedef __attribute__((ext_vector_type(8)))  float  v8f;

#define B_   128
#define S_   512
#define I_   256
#define H_   512
#define G4_  2048   /* 4*H */
#define NWG  32     /* H/16 column-owning workgroups */
#define TPB  256    /* 8 waves (wave32) */

#define K0_   (I_ + H_)      /* 768  */
#define K1_   (H_ + H_)      /* 1024 */
#define LDS0S (K0_ + 8)      /* padded LDS row stride (elements): 776  -> 1552B, 388dw, 388%64=4 */
#define LDS1S (K1_ + 8)      /* 1032 -> 2064B, 516dw, 516%64=4 */

/* ---------------- workspace layout (bytes) ---------------- */
#define OFF_BAR   0u
#define OFF_H0    256u                            /* bf16 ping+pong: 2*128KB */
#define OFF_H1    (OFF_H0 + 2u*131072u)           /* bf16 ping+pong: 2*128KB */
#define OFF_C0    (OFF_H1 + 2u*131072u)           /* f32 256KB */
#define OFF_C1    (OFF_C0 + 262144u)              /* f32 256KB */
#define MEMSET_BYTES (OFF_C1 + 262144u)           /* zero bar+h+c each call  */
#define OFF_H1F   MEMSET_BYTES                    /* f32 256KB (last h1)     */
#define OFF_B0    (OFF_H1F + 262144u)             /* f32 8KB combined bias   */
#define OFF_B1    (OFF_B0 + 8192u)
#define OFF_W0    (OFF_B1 + 8192u)                /* bf16 2048x768  */
#define OFF_W1    (OFF_W0 + 3145728u)             /* bf16 2048x1024 */
#define OFF_XT    (OFF_W1 + 4194304u)             /* bf16 [S,B,I] time-major */

__device__ __forceinline__ float sigm(float x) { return 1.0f / (1.0f + __expf(-x)); }

/* 32B LDS fragment via two 16B-aligned halves -> 2x ds_load_b128 */
__device__ __forceinline__ v16bf ld_frag_lds(const __bf16* p) {
  union { v16bf v; v8bf h[2]; } u;
  u.h[0] = *(const v8bf*)(p);
  u.h[1] = *(const v8bf*)(p + 8);
  return u.v;
}

#define WMMA_BF16(A, Bv, C) __builtin_amdgcn_wmma_f32_16x16x32_bf16( \
    false, (A), false, (Bv), (short)0, (C), false, false)

/* ---- concat [W_ih | W_hh] row-major -> bf16, K = Iin + H ---- */
__global__ void k_cvt_wcat(__bf16* __restrict__ dst, const float* __restrict__ Wih,
                           const float* __restrict__ Whh, int Iin) {
  const int K = Iin + H_;
  const size_t total = (size_t)G4_ * (size_t)K;
  for (size_t idx = blockIdx.x * (size_t)blockDim.x + threadIdx.x; idx < total;
       idx += (size_t)gridDim.x * blockDim.x) {
    int n = (int)(idx / K);
    int k = (int)(idx % K);
    float v = (k < Iin) ? Wih[(size_t)n * Iin + k] : Whh[(size_t)n * H_ + (k - Iin)];
    dst[idx] = (__bf16)v;
  }
}

__global__ void k_bias(float* __restrict__ dst, const float* __restrict__ a,
                       const float* __restrict__ b) {
  int i = blockIdx.x * blockDim.x + threadIdx.x;
  if (i < G4_) dst[i] = a[i] + b[i];
}

/* x [B,S,I] f32 -> xT [S,B,I] bf16 */
__global__ void k_cvt_x(__bf16* __restrict__ dst, const float* __restrict__ x) {
  const size_t total = (size_t)S_ * B_ * I_;
  for (size_t idx = blockIdx.x * (size_t)blockDim.x + threadIdx.x; idx < total;
       idx += (size_t)gridDim.x * blockDim.x) {
    size_t i  = idx % I_;
    size_t sb = idx / I_;
    size_t b  = sb % B_;
    size_t s  = sb / B_;
    dst[idx] = (__bf16)x[(b * S_ + s) * I_ + i];
  }
}

/* ---- grid barrier: agent-scope release/acquire + s_sleep spin ---- */
__device__ __forceinline__ void grid_barrier(unsigned* cnt, unsigned* gen) {
  __syncthreads();
  if (threadIdx.x == 0) {
    unsigned my = __hip_atomic_load(gen, __ATOMIC_RELAXED, __HIP_MEMORY_SCOPE_AGENT);
    __threadfence();
    unsigned prev = __hip_atomic_fetch_add(cnt, 1u, __ATOMIC_ACQ_REL, __HIP_MEMORY_SCOPE_AGENT);
    if (prev == NWG - 1) {
      __hip_atomic_store(cnt, 0u, __ATOMIC_RELAXED, __HIP_MEMORY_SCOPE_AGENT);
      __hip_atomic_fetch_add(gen, 1u, __ATOMIC_RELEASE, __HIP_MEMORY_SCOPE_AGENT);
    } else {
      while (__hip_atomic_load(gen, __ATOMIC_ACQUIRE, __HIP_MEMORY_SCOPE_AGENT) == my)
        __builtin_amdgcn_s_sleep(2);
    }
  }
  __syncthreads();
}

/* cooperative copy of this WG's 64 weight rows (bf16, width K) into padded LDS */
__device__ __forceinline__ void load_w_lds(__bf16* dstL, const __bf16* __restrict__ Wcat,
                                           int K, int strideEl, int col0) {
  const int r = threadIdx.x >> 2;          /* local row 0..63  (g = r/16, lm = r%16) */
  const int q = threadIdx.x & 3;           /* quarter of the row */
  const int n = (r >> 4) * H_ + col0 + (r & 15);
  const int qe = K >> 2;                   /* elements per quarter */
  const uint4* src = (const uint4*)(Wcat + (size_t)n * K + q * qe);
  uint4*       dst = (uint4*)(dstL + (size_t)r * strideEl + q * qe);
  const int nv = K >> 5;                   /* uint4 chunks per quarter = K/32 */
#pragma unroll 4
  for (int i = 0; i < nv; ++i) dst[i] = src[i];
}

/* One LSTM layer step for this wave's 16x64 gate tile.
 * A = [Aext (kExt cols) | Ah (H cols)] row-major bf16 in global.
 * B fragments come from LDS (this WG's 64 weight rows, padded stride). */
__device__ __forceinline__ void lstm_phase(
    const __bf16* __restrict__ Aext, int ldaExt, int kExt,
    const __bf16* __restrict__ Ah,
    const __bf16* Wl, int strideEl,
    const float*  __restrict__ bias,
    float*  __restrict__ cbuf,
    __bf16* __restrict__ hout,
    float*  __restrict__ hf32,
    int col0, int m0, int lane) {
  const int half = lane >> 4;
  const int lm   = lane & 15;
  const int boff = half * 16;

  v8f acc0, acc1, acc2, acc3;
  {
    float b0v = bias[0 * H_ + col0 + lm];
    float b1v = bias[1 * H_ + col0 + lm];
    float b2v = bias[2 * H_ + col0 + lm];
    float b3v = bias[3 * H_ + col0 + lm];
    acc0 = (v8f){b0v, b0v, b0v, b0v, b0v, b0v, b0v, b0v};
    acc1 = (v8f){b1v, b1v, b1v, b1v, b1v, b1v, b1v, b1v};
    acc2 = (v8f){b2v, b2v, b2v, b2v, b2v, b2v, b2v, b2v};
    acc3 = (v8f){b3v, b3v, b3v, b3v, b3v, b3v, b3v, b3v};
  }

  const __bf16* w0 = Wl + (size_t)( 0 + lm) * strideEl + boff;
  const __bf16* w1 = Wl + (size_t)(16 + lm) * strideEl + boff;
  const __bf16* w2 = Wl + (size_t)(32 + lm) * strideEl + boff;
  const __bf16* w3 = Wl + (size_t)(48 + lm) * strideEl + boff;

  /* --- x / layer-input part of K --- */
  {
    const __bf16* arow = Aext + (size_t)(m0 + lm) * ldaExt + boff;
#pragma unroll 2
    for (int kt = 0; kt < kExt; kt += 32) {
      v16bf a = *(const v16bf*)(arow + kt);
      acc0 = WMMA_BF16(a, ld_frag_lds(w0 + kt), acc0);
      acc1 = WMMA_BF16(a, ld_frag_lds(w1 + kt), acc1);
      acc2 = WMMA_BF16(a, ld_frag_lds(w2 + kt), acc2);
      acc3 = WMMA_BF16(a, ld_frag_lds(w3 + kt), acc3);
    }
  }
  /* --- recurrent part of K --- */
  {
    const __bf16* hrow = Ah + (size_t)(m0 + lm) * H_ + boff;
#pragma unroll 2
    for (int kt = 0; kt < H_; kt += 32) {
      v16bf a = *(const v16bf*)(hrow + kt);
      acc0 = WMMA_BF16(a, ld_frag_lds(w0 + kExt + kt), acc0);
      acc1 = WMMA_BF16(a, ld_frag_lds(w1 + kExt + kt), acc1);
      acc2 = WMMA_BF16(a, ld_frag_lds(w2 + kExt + kt), acc2);
      acc3 = WMMA_BF16(a, ld_frag_lds(w3 + kExt + kt), acc3);
    }
  }

  /* C/D layout: element j of lane L -> (m = m0 + 8*half + j, n = col0 + lm) */
  const int col = col0 + lm;
#pragma unroll
  for (int j = 0; j < 8; ++j) {
    int row = m0 + half * 8 + j;
    size_t idx = (size_t)row * H_ + col;
    float iv = sigm(acc0[j]);
    float fv = sigm(acc1[j]);
    float gv = tanhf(acc2[j]);
    float ov = sigm(acc3[j]);
    float c  = fv * cbuf[idx] + iv * gv;
    cbuf[idx] = c;
    float h = ov * tanhf(c);
    hout[idx] = (__bf16)h;
    if (hf32) hf32[idx] = h;
  }
}

__global__ void __launch_bounds__(TPB, 1)
k_lstm(const __bf16* __restrict__ xT,
       const __bf16* __restrict__ W0, const __bf16* __restrict__ W1,
       const float*  __restrict__ b0, const float*  __restrict__ b1,
       __bf16* __restrict__ h0, __bf16* __restrict__ h1,
       float*  __restrict__ c0, float*  __restrict__ c1,
       float*  __restrict__ h1f, unsigned* __restrict__ bar) {
  /* this WG's 64 weight rows for both layers, padded rows: 231,424 B of LDS */
  __shared__ __align__(16) __bf16 ldsW[64 * LDS0S + 64 * LDS1S];
  __bf16* lW0 = ldsW;
  __bf16* lW1 = ldsW + 64 * LDS0S;

  const int lane = threadIdx.x & 31;
  const int m0   = (threadIdx.x >> 5) * 16;   /* wave -> M tile  */
  const int col0 = blockIdx.x * 16;           /* WG   -> h cols  */
  unsigned* cnt = bar;
  unsigned* gen = bar + 1;
  const size_t HB = (size_t)B_ * H_;

  load_w_lds(lW0, W0, K0_, LDS0S, col0);
  load_w_lds(lW1, W1, K1_, LDS1S, col0);
  __syncthreads();

  for (int s = 0; s < S_; ++s) {
    /* keep next timestep's x panel (64KB) hot in L2 ahead of the scan */
    if (s + 1 < S_)
      __builtin_prefetch((const char*)(xT + (size_t)(s + 1) * B_ * I_) + threadIdx.x * 256, 0, 1);

    const __bf16* h0r = h0 + (size_t)(s & 1) * HB;
    __bf16*       h0w = h0 + (size_t)((s & 1) ^ 1) * HB;
    const __bf16* h1r = h1 + (size_t)(s & 1) * HB;
    __bf16*       h1w = h1 + (size_t)((s & 1) ^ 1) * HB;

    /* layer 0: A = [x_s | h0_prev], K = 768 */
    lstm_phase(xT + (size_t)s * B_ * I_, I_, I_, h0r, lW0, LDS0S, b0,
               c0, h0w, nullptr, col0, m0, lane);
    grid_barrier(cnt, gen);

    /* layer 1: A = [h0_new | h1_prev], K = 1024 */
    lstm_phase(h0w, H_, H_, h1r, lW1, LDS1S, b1,
               c1, h1w, (s == S_ - 1) ? h1f : nullptr, col0, m0, lane);
    grid_barrier(cnt, gen);
  }
}

/* tiny FC: out[b,j] = h1[b,:].fc_w[j,:] + fc_b[j]; 256 threads = 128x2 */
__global__ void k_fc(const float* __restrict__ h, const float* __restrict__ w,
                     const float* __restrict__ bias, float* __restrict__ out) {
  int t = threadIdx.x;
  int b = t >> 1, j = t & 1;
  float acc = 0.f;
  for (int k = 0; k < H_; ++k) acc += h[b * H_ + k] * w[j * H_ + k];
  out[b * 2 + j] = acc + bias[j];
}

extern "C" void kernel_launch(void* const* d_in, const int* in_sizes, int n_in,
                              void* d_out, int out_size, void* d_ws, size_t ws_size,
                              hipStream_t stream) {
  (void)in_sizes; (void)n_in; (void)out_size; (void)ws_size;
  const float* x    = (const float*)d_in[0];
  const float* Wih0 = (const float*)d_in[1];
  const float* Whh0 = (const float*)d_in[2];
  const float* bih0 = (const float*)d_in[3];
  const float* bhh0 = (const float*)d_in[4];
  const float* Wih1 = (const float*)d_in[5];
  const float* Whh1 = (const float*)d_in[6];
  const float* bih1 = (const float*)d_in[7];
  const float* bhh1 = (const float*)d_in[8];
  const float* fcw  = (const float*)d_in[9];
  const float* fcb  = (const float*)d_in[10];

  char* ws = (char*)d_ws;
  unsigned* bar = (unsigned*)(ws + OFF_BAR);
  __bf16* h0   = (__bf16*)(ws + OFF_H0);
  __bf16* h1   = (__bf16*)(ws + OFF_H1);
  float*  c0   = (float*)(ws + OFF_C0);
  float*  c1   = (float*)(ws + OFF_C1);
  float*  h1f  = (float*)(ws + OFF_H1F);
  float*  bs0  = (float*)(ws + OFF_B0);
  float*  bs1  = (float*)(ws + OFF_B1);
  __bf16* W0   = (__bf16*)(ws + OFF_W0);
  __bf16* W1   = (__bf16*)(ws + OFF_W1);
  __bf16* xT   = (__bf16*)(ws + OFF_XT);

  /* zero barrier + h/c state every call (graph-capture safe) */
  hipMemsetAsync(ws, 0, MEMSET_BYTES, stream);

  k_cvt_wcat<<<1024, 256, 0, stream>>>(W0, Wih0, Whh0, I_);
  k_cvt_wcat<<<1024, 256, 0, stream>>>(W1, Wih1, Whh1, H_);
  k_bias<<<(G4_ + 255) / 256, 256, 0, stream>>>(bs0, bih0, bhh0);
  k_bias<<<(G4_ + 255) / 256, 256, 0, stream>>>(bs1, bih1, bhh1);
  k_cvt_x<<<2048, 256, 0, stream>>>(xT, x);

  k_lstm<<<NWG, TPB, 0, stream>>>(xT, W0, W1, bs0, bs1, h0, h1, c0, c1, h1f, bar);

  k_fc<<<1, 256, 0, stream>>>(h1f, fcw, fcb, (float*)d_out);
}
